// SpatialSelfAttention_49323404427658
// MI455X (gfx1250) — matrix operand order repaired and checked by reading, not compiled, not run on hardware
//
#include <hip/hip_runtime.h>
#include <hip/hip_bf16.h>

typedef __attribute__((ext_vector_type(16))) _Float16 v16h;
typedef __attribute__((ext_vector_type(8)))  _Float16 v8h;
typedef __attribute__((ext_vector_type(8)))  float    v8f;
typedef __attribute__((ext_vector_type(4)))  float    v4f;

#define BB 4
#define NN 4096
#define CC 128
#define LDSW 136   // padded row stride (halves) to spread LDS banks

union H16 { v16h v; v8h h2[2]; };
union H8I { v8h h; int i[4]; };

// ---------------------------------------------------------------------------
// Kernel 0: convert the three [C_in][C_out] f32 weights into transposed f16
// Wt[j][c] = W[c][j] so WMMA B-operands read contiguous rows.
// ---------------------------------------------------------------------------
__global__ void weights_cvt_kernel(const float* __restrict__ wq,
                                   const float* __restrict__ wk,
                                   const float* __restrict__ wv,
                                   _Float16* __restrict__ wtq,
                                   _Float16* __restrict__ wtk,
                                   _Float16* __restrict__ wtv) {
  int idx = blockIdx.x * blockDim.x + threadIdx.x;
  if (idx >= 3 * CC * CC) return;
  int sel = idx / (CC * CC);
  int r   = idx % (CC * CC);
  int j = r / CC, c = r % CC;
  const float* w = (sel == 0) ? wq : (sel == 1) ? wk : wv;
  _Float16*   wt = (sel == 0) ? wtq : (sel == 1) ? wtk : wtv;
  wt[r] = (_Float16)w[c * CC + j];
}

// ---------------------------------------------------------------------------
// Kernel 1: QKV projection via WMMA.  Each wave: 16 rows x 128 cols x 3 mats.
// Q,K stored row-major f16 [n][c] (transposed through wave-private LDS so the
// global stores are coalesced b128); V stored transposed f16 [c][n] directly.
// ---------------------------------------------------------------------------
template <bool TRANS>
__device__ __forceinline__ void proj_one(const v16h a[4],
                                         const _Float16* __restrict__ wt,
                                         const float* __restrict__ bias,
                                         _Float16* __restrict__ dst,
                                         _Float16* lt,   // wave-private LDS tile
                                         int ntile, int lane) {
  const int h = lane >> 4, lr = lane & 15;
#pragma unroll
  for (int jt = 0; jt < 8; ++jt) {
    const int j = jt * 16 + lr;
    const _Float16* wr = wt + (size_t)j * CC;
    v8f acc = {};
#pragma unroll
    for (int kk = 0; kk < 4; ++kk) {
      // B operand: lane holds column j; element e <-> c = kk*32 + h*16 + e
      H16 b;
      b.h2[0] = *(const v8h*)(wr + kk * 32 + h * 16);
      b.h2[1] = *(const v8h*)(wr + kk * 32 + h * 16 + 8);
      acc = __builtin_amdgcn_wmma_f32_16x16x32_f16(false, a[kk], false, b.v,
                                                   (short)0, acc, false, false);
    }
    const float bj = bias[j];
    if (TRANS) {
      // V: D layout already contiguous in n for fixed channel j -> direct b128
      v8h ph;
#pragma unroll
      for (int r = 0; r < 8; ++r) ph[r] = (_Float16)(acc[r] + bj);
      *(v8h*)(dst + (size_t)j * NN + ntile + h * 8) = ph;
    } else {
      // Q/K: scatter the transpose into LDS (DS ops are in-order, cheap)
#pragma unroll
      for (int r = 0; r < 8; ++r)
        lt[(h * 8 + r) * LDSW + j] = (_Float16)(acc[r] + bj);
    }
  }
  if (!TRANS) {
    // gather contiguous rows back and emit coalesced 16-B global stores
#pragma unroll
    for (int i = 0; i < 4; ++i) {
      v8h row = *(const v8h*)(lt + lr * LDSW + h * 64 + i * 16);
      *(v8h*)(dst + (size_t)(ntile + lr) * CC + h * 64 + i * 16) = row;
    }
  }
}

__global__ void proj_kernel(const float* __restrict__ x,
                            const _Float16* __restrict__ wtq, const float* __restrict__ bq,
                            const _Float16* __restrict__ wtk, const float* __restrict__ bk,
                            const _Float16* __restrict__ wtv, const float* __restrict__ bv,
                            _Float16* __restrict__ Qh, _Float16* __restrict__ Kh,
                            _Float16* __restrict__ Vt) {
  __shared__ _Float16 lds_t[4][16 * LDSW];
  const int lane = threadIdx.x & 31;
  const int wid  = threadIdx.x >> 5;
  const int h = lane >> 4, lr = lane & 15;
  const int batch = blockIdx.x >> 6;                 // 64 blocks per batch
  const int ntile = ((blockIdx.x & 63) << 6) + (wid << 4);
  const int row   = ntile + lr;

  // A operand (x rows, f32 -> f16): element e <-> c = kk*32 + h*16 + e
  const float* xrow = x + ((size_t)batch * NN + row) * CC;
  v16h a[4];
#pragma unroll
  for (int kk = 0; kk < 4; ++kk) {
    const int c0 = kk * 32 + h * 16;
    v4f f0 = *(const v4f*)(xrow + c0);
    v4f f1 = *(const v4f*)(xrow + c0 + 4);
    v4f f2 = *(const v4f*)(xrow + c0 + 8);
    v4f f3 = *(const v4f*)(xrow + c0 + 12);
    H16 u;
#pragma unroll
    for (int e = 0; e < 4; ++e) {
      u.v[e]      = (_Float16)f0[e];
      u.v[4 + e]  = (_Float16)f1[e];
      u.v[8 + e]  = (_Float16)f2[e];
      u.v[12 + e] = (_Float16)f3[e];
    }
    a[kk] = u.v;
  }

  _Float16* lt = &lds_t[wid][0];
  _Float16* Qb = Qh + (size_t)batch * NN * CC;
  _Float16* Kb = Kh + (size_t)batch * NN * CC;
  _Float16* Vb = Vt + (size_t)batch * CC * NN;
  proj_one<false>(a, wtq, bq, Qb, lt, ntile, lane);
  proj_one<false>(a, wtk, bk, Kb, lt, ntile, lane);
  proj_one<true >(a, wtv, bv, Vb, lt, ntile, lane);
}

// ---------------------------------------------------------------------------
// Kernel 2: fused flash attention.  One wave = 16 query rows; iterate keys in
// chunks of 64 (amortizes the online-softmax rescale).  S^T = K * Q^T so the
// softmax reductions are lane-local per query (one shfl_xor(16) combine);
// O^T = V^T * P^T accumulated in 8 f32 tiles.
// ---------------------------------------------------------------------------
__global__ void flash_kernel(const _Float16* __restrict__ Qh,
                             const _Float16* __restrict__ Kh,
                             const _Float16* __restrict__ Vt,
                             float* __restrict__ out) {
  const int lane = threadIdx.x & 31;
  const int wid  = threadIdx.x >> 5;
  const int h = lane >> 4, lr = lane & 15;
  const int batch = blockIdx.x >> 6;
  const int qtile = ((blockIdx.x & 63) << 6) + (wid << 4);
  const int qrow  = qtile + lr;

  const _Float16* Qb = Qh + (size_t)batch * NN * CC;
  const _Float16* Kb = Kh + (size_t)batch * NN * CC;
  const _Float16* Vb = Vt + (size_t)batch * CC * NN;

  // Q as B operand of S^T: lane holds column q = lr; element e <-> c = kk*32 + h*16 + e
  v16h qreg[4];
  {
    const _Float16* qp = Qb + (size_t)qrow * CC;
#pragma unroll
    for (int kk = 0; kk < 4; ++kk) {
      H16 u;
      u.h2[0] = *(const v8h*)(qp + kk * 32 + h * 16);
      u.h2[1] = *(const v8h*)(qp + kk * 32 + h * 16 + 8);
      qreg[kk] = u.v;
    }
  }

  v8f o[8] = {};                       // O^T accumulators, 8 d-tiles of 16 ch
  float run_max = -3.0e38f;            // in scaled-log2 domain
  float lsum = 0.0f;
  const float SCL2E = 0.08838834764831845f * 1.4426950408889634f; // 1/sqrt(C)*log2(e)

  for (int mb = 0; mb < NN; mb += 64) {
    // ---- S^T tiles for 4 subtiles of 16 keys ----
    v8f st[4];
#pragma unroll
    for (int t = 0; t < 4; ++t) {
      const _Float16* kp = Kb + (size_t)(mb + t * 16 + lr) * CC;
      v8f acc = {};
#pragma unroll
      for (int kk = 0; kk < 4; ++kk) {
        const int c0 = kk * 32 + h * 16;
        H16 ka;
        ka.h2[0] = *(const v8h*)(kp + c0);
        ka.h2[1] = *(const v8h*)(kp + c0 + 8);
        acc = __builtin_amdgcn_wmma_f32_16x16x32_f16(false, ka.v, false, qreg[kk],
                                                     (short)0, acc, false, false);
      }
      st[t] = acc;
      __builtin_prefetch(kp + (size_t)64 * CC + h * 64, 0, 1);  // next chunk hint
    }

    // ---- online softmax (base-2, scale folded into one FMA per score) ----
    float cmax = -3.0e38f;
#pragma unroll
    for (int t = 0; t < 4; ++t)
#pragma unroll
      for (int r = 0; r < 8; ++r) cmax = fmaxf(cmax, st[t][r]);
    cmax = fmaxf(cmax, __shfl_xor(cmax, 16, 32));
    const float nm   = fmaxf(run_max, cmax * SCL2E);
    const float corr = exp2f(run_max - nm);
    float csum = 0.0f;
    H8I a[4];
#pragma unroll
    for (int t = 0; t < 4; ++t) {
#pragma unroll
      for (int r = 0; r < 8; ++r) {
        const float p = exp2f(fmaf(st[t][r], SCL2E, -nm));
        csum += p;
        a[t].h[r] = (_Float16)p;
      }
    }
    csum += __shfl_xor(csum, 16, 32);
    lsum = lsum * corr + csum;
    run_max = nm;

    // ---- build the two P^T B-operands (half-wave dword exchange) ----
    H8I xw[4];
#pragma unroll
    for (int t = 0; t < 4; ++t)
#pragma unroll
      for (int i = 0; i < 4; ++i) xw[t].i[i] = __shfl_xor(a[t].i[i], 16, 32);
    H16 p01, p23;
    p01.h2[0] = h ? xw[1].h : a[0].h;   // keys mb    + [h*16,   h*16+8)
    p01.h2[1] = h ? a[1].h : xw[0].h;   // keys mb    + [h*16+8, h*16+16)
    p23.h2[0] = h ? xw[3].h : a[2].h;   // keys mb+32 + ...
    p23.h2[1] = h ? a[3].h : xw[2].h;

    // ---- rescale accumulators once per 64 keys, then O^T += V^T P^T ----
#pragma unroll
    for (int dt = 0; dt < 8; ++dt) {
#pragma unroll
      for (int r = 0; r < 8; ++r) o[dt][r] *= corr;
    }
#pragma unroll
    for (int dt = 0; dt < 8; ++dt) {
      const _Float16* vp = Vb + (size_t)(dt * 16 + lr) * NN + mb;
      H16 va;
      va.h2[0] = *(const v8h*)(vp + h * 16);
      va.h2[1] = *(const v8h*)(vp + 8 + h * 16);
      o[dt] = __builtin_amdgcn_wmma_f32_16x16x32_f16(false, va.v, false, p01.v,
                                                     (short)0, o[dt], false, false);
      va.h2[0] = *(const v8h*)(vp + 32 + h * 16);
      va.h2[1] = *(const v8h*)(vp + 40 + h * 16);
      o[dt] = __builtin_amdgcn_wmma_f32_16x16x32_f16(false, va.v, false, p23.v,
                                                     (short)0, o[dt], false, false);
      __builtin_prefetch(vp + 64, 0, 1);               // next chunk hint
    }
  }

  // ---- finalize: out[q][d] = O^T[d][q] / lsum ----
  const float inv = 1.0f / lsum;
  float* orow = out + ((size_t)batch * NN + qrow) * CC;
#pragma unroll
  for (int dt = 0; dt < 8; ++dt) {
    v4f w0, w1;
#pragma unroll
    for (int r = 0; r < 4; ++r) {
      w0[r] = o[dt][r] * inv;
      w1[r] = o[dt][4 + r] * inv;
    }
    *(v4f*)(orow + dt * 16 + h * 8)     = w0;
    *(v4f*)(orow + dt * 16 + h * 8 + 4) = w1;
  }
}

// ---------------------------------------------------------------------------
extern "C" void kernel_launch(void* const* d_in, const int* in_sizes, int n_in,
                              void* d_out, int out_size, void* d_ws, size_t ws_size,
                              hipStream_t stream) {
  const float* x  = (const float*)d_in[0];
  const float* wq = (const float*)d_in[1];
  const float* bq = (const float*)d_in[2];
  const float* wk = (const float*)d_in[3];
  const float* bk = (const float*)d_in[4];
  const float* wv = (const float*)d_in[5];
  const float* bv = (const float*)d_in[6];
  (void)in_sizes; (void)n_in; (void)out_size; (void)ws_size;

  // Workspace layout (f16 elements)
  _Float16* ws  = (_Float16*)d_ws;
  _Float16* wtq = ws;                               // 128*128
  _Float16* wtk = wtq + CC * CC;
  _Float16* wtv = wtk + CC * CC;
  _Float16* Qh  = wtv + CC * CC;                    // B*N*C
  _Float16* Kh  = Qh + (size_t)BB * NN * CC;
  _Float16* Vt  = Kh + (size_t)BB * NN * CC;        // transposed [B][C][N]

  weights_cvt_kernel<<<(3 * CC * CC + 255) / 256, 256, 0, stream>>>(
      wq, wk, wv, wtq, wtk, wtv);

  // 4 batches * 64 row-blocks of 64 rows; 4 waves (16 rows each) per block
  proj_kernel<<<BB * 64, 128, 0, stream>>>(x, wtq, bq, wtk, bk, wtv, bv,
                                           Qh, Kh, Vt);

  flash_kernel<<<BB * 64, 128, 0, stream>>>(Qh, Kh, Vt, (float*)d_out);
}
